// QueryEncoderExpand_46643344834806
// MI455X (gfx1250) — compile-verified, matrix operand order checked by reading, not run on hardware
//
#include <hip/hip_runtime.h>
#include <hip/hip_bf16.h>

// ---------------- problem constants (match reference) ----------------
#define B_    4
#define W_    2000
#define I_    64
#define S_    64
#define T_    24
#define U_    32
#define US_   2
#define PER_  (W_ + I_ + S_ + U_ + 1)     // 2161
#define NN_   (B_ * PER_)                 // 8644 nodes
#define NNPAD 8704                        // 68 * 128 (padded rows for GEMM A loads)
#define V_    32000
#define H_    4
#define D_    256
#define HD_   (H_ * D_)                   // 1024
#define OUT_  512
#define K300P 320                         // 300 padded to 32-multiple

typedef __attribute__((ext_vector_type(16))) _Float16 v16h;
typedef __attribute__((ext_vector_type(8)))  _Float16 v8h;
typedef __attribute__((ext_vector_type(8)))  float    v8f;

// ---- CDNA5 async global->LDS copy (ISA 15.18, ASYNCcnt-tracked) ----
static __device__ __forceinline__ void async_copy_b128(void* lds, const void* g)
{
    unsigned l = (unsigned)(unsigned long long)lds;          // LDS byte offset
    unsigned long long ga = (unsigned long long)g;           // global VA
    asm volatile("global_load_async_to_lds_b128 %0, %1, off"
                 :: "v"(l), "v"(ga) : "memory");
}
static __device__ __forceinline__ void wait_async0()
{
    asm volatile("s_wait_asynccnt 0x0" ::: "memory");
}
static __device__ __forceinline__ void wait_async3()
{
    // <=3 outstanding: the previous tile's 3 copies (per thread) have completed
    asm volatile("s_wait_asynccnt 0x3" ::: "memory");
}

// Load one 16x32 fragment row-slice (per-lane two 16B chunks) from an LDS row.
static __device__ __forceinline__ v16h load_frag(const _Float16* rowp, int kh)
{
    v8h lo = *(const v8h*)(rowp + kh);
    v8h hi = *(const v8h*)(rowp + kh + 16);
    v16h r;
#pragma unroll
    for (int i = 0; i < 8; ++i) { r[i] = lo[i]; r[8 + i] = hi[i]; }
    return r;
}

// ===================================================================
// Tiled WMMA GEMM (all-f16 data path, double-buffered async staging):
//   C[M,Ncols](f16) = A[M,K](f16) @ Bt[Ncols,K](f16, pre-transposed) (+bias)
// LDS tiles filled via global_load_async_to_lds_b128 one K-step ahead;
// f32 accumulate. Block: 256 threads = 8 waves (4x2); block tile 128x64;
// each wave computes a 32x32 tile = 2x2 WMMA fragments (4 wmma / K-step).
// A buffers are row-padded so A-tile loads need no bounds checks.
// ===================================================================
__global__ void __launch_bounds__(256)
gemm_f16_wmma(const _Float16* __restrict__ A, int lda,
              const _Float16* __restrict__ Bt,   // [Ncols][K]
              const float* __restrict__ bias,    // [Ncols] or nullptr
              _Float16* __restrict__ C, int ldc,
              const int* __restrict__ rowmap,    // remap of output rows or nullptr
              int M, int Ncols, int K)
{
    __shared__ _Float16 As[2][128][32];  // 16 KB
    __shared__ _Float16 Bs[2][64][32];   //  8 KB

    const int tid  = threadIdx.x;
    const int m0   = blockIdx.x * 128;
    const int n0   = blockIdx.y * 64;
    const int wave = tid >> 5;
    const int lane = tid & 31;
    const int wr   = wave >> 1;       // 0..3 -> rows m0 + wr*32
    const int wc   = wave & 1;        // 0..1 -> cols n0 + wc*32
    const int lrow = lane & 15;
    const int kh   = (lane >> 4) * 8;

    const int row = tid >> 2;         // 0..63
    const int q8  = (tid & 3) * 8;    // 0,8,16,24 (halves): 16B chunk

    const _Float16* aRow0 = A  + (size_t)(m0 + row)      * lda + q8;
    const _Float16* aRow1 = A  + (size_t)(m0 + row + 64) * lda + q8;
    const _Float16* bRow  = Bt + (size_t)(n0 + row)      * K   + q8;

    v8f acc00 = {}, acc01 = {}, acc10 = {}, acc11 = {};

    // prologue: stage tile 0 into buffer 0
    async_copy_b128(&As[0][row][q8],      aRow0);
    async_copy_b128(&As[0][row + 64][q8], aRow1);
    async_copy_b128(&Bs[0][row][q8],      bRow);

    const int nIter = K >> 5;
    for (int i = 0; i < nIter; ++i) {
        const int cur = i & 1;
        if (i + 1 < nIter) {
            const int nxt = cur ^ 1;
            const int kn  = (i + 1) << 5;
            async_copy_b128(&As[nxt][row][q8],      aRow0 + kn);
            async_copy_b128(&As[nxt][row + 64][q8], aRow1 + kn);
            async_copy_b128(&Bs[nxt][row][q8],      bRow  + kn);
            wait_async3();
        } else {
            wait_async0();
        }
        __syncthreads();

        // ---- 2x2 fragments, 4 WMMAs with A/B reuse ----
        v16h a0 = load_frag(&As[cur][wr * 32 +      lrow][0], kh);
        v16h a1 = load_frag(&As[cur][wr * 32 + 16 + lrow][0], kh);
        v16h b0 = load_frag(&Bs[cur][wc * 32 +      lrow][0], kh);
        v16h b1 = load_frag(&Bs[cur][wc * 32 + 16 + lrow][0], kh);
        acc00 = __builtin_amdgcn_wmma_f32_16x16x32_f16(false, a0, false, b0, (short)0, acc00, false, false);
        acc01 = __builtin_amdgcn_wmma_f32_16x16x32_f16(false, a0, false, b1, (short)0, acc01, false, false);
        acc10 = __builtin_amdgcn_wmma_f32_16x16x32_f16(false, a1, false, b0, (short)0, acc10, false, false);
        acc11 = __builtin_amdgcn_wmma_f32_16x16x32_f16(false, a1, false, b1, (short)0, acc11, false, false);
        __syncthreads();   // protects buffer reuse two iterations ahead
    }

    // ---- store (f16): VGPR i -> row (lane<16 ? i : i+8), col = lane&15 ----
    const int rbase = (lane >> 4) * 8;
    const int col0  = n0 + wc * 32 + (lane & 15);
    const int col1  = col0 + 16;
    const float bv0 = bias ? bias[col0] : 0.0f;
    const float bv1 = bias ? bias[col1] : 0.0f;
#pragma unroll
    for (int i = 0; i < 8; ++i) {
        int gm0 = m0 + wr * 32 + rbase + i;
        int gm1 = gm0 + 16;
        if (gm0 < M) {
            int orow = rowmap ? rowmap[gm0] : gm0;
            C[(size_t)orow * ldc + col0] = (_Float16)(acc00[i] + bv0);
            C[(size_t)orow * ldc + col1] = (_Float16)(acc01[i] + bv1);
        }
        if (gm1 < M) {
            int orow = rowmap ? rowmap[gm1] : gm1;
            C[(size_t)orow * ldc + col0] = (_Float16)(acc10[i] + bv0);
            C[(size_t)orow * ldc + col1] = (_Float16)(acc11[i] + bv1);
        }
    }
}

// ===================================================================
// Weight convert + transpose:  W[Kreal,N] fp32  ->  Wt[N][Kpad] f16
// ===================================================================
__global__ void conv_weights(const float* __restrict__ W, _Float16* __restrict__ Wt,
                             int Kreal, int Kpad, int N)
{
    int idx = blockIdx.x * blockDim.x + threadIdx.x;
    if (idx >= N * Kpad) return;
    int n = idx / Kpad, k = idx % Kpad;
    Wt[idx] = (k < Kreal) ? (_Float16)W[(size_t)k * N + n] : (_Float16)0.0f;
}

__global__ void conv_f32_to_f16(const float* __restrict__ src, _Float16* __restrict__ dst, int n)
{
    int i = blockIdx.x * blockDim.x + threadIdx.x;
    if (i < n) dst[i] = (_Float16)src[i];
}

// ===================================================================
// Feature construction (writes f16 node matrix [NNPAD, 320])
// ===================================================================
__global__ void build_words(const int* __restrict__ all_words,
                            const float* __restrict__ we,
                            _Float16* __restrict__ nodeF)
{
    int idx = blockIdx.x * blockDim.x + threadIdx.x;
    const int total = B_ * W_ * K300P;
    if (idx >= total) return;
    int row = idx / K300P, c = idx % K300P;
    int b = row / W_, w = row % W_;
    float v = 0.0f;
    if (c < 300) v = we[(size_t)all_words[row] * 300 + c];
    nodeF[(size_t)(b * PER_ + w) * K300P + c] = (_Float16)v;
}

__global__ void build_sent(const int* __restrict__ sents,
                           const float* __restrict__ smask,
                           const float* __restrict__ we,
                           float* __restrict__ sent300,
                           _Float16* __restrict__ nodeF)
{
    int bs = blockIdx.x;                   // 0..B*S-1
    int b = bs / S_, s = bs % S_;
    int c = threadIdx.x;                   // 320 threads
    const int*   sw = sents + (size_t)bs * T_;
    const float* mk = smask + (size_t)bs * T_;
    float cnt = 0.0f;
    for (int t = 0; t < T_; ++t) cnt += mk[t];
    float inv = 1.0f / ((cnt == 0.0f) ? 1.0f : cnt);
    float v = 0.0f;
    if (c < 300) {
        float acc = 0.0f;
        for (int t = 0; t < T_; ++t) acc += mk[t] * we[(size_t)sw[t] * 300 + c];
        v = acc * inv;
        sent300[(size_t)bs * 300 + c] = v;
    }
    nodeF[(size_t)(b * PER_ + W_ + I_ + s) * K300P + c] = (_Float16)v;
}

__global__ void build_utt(const int* __restrict__ utts,
                          const float* __restrict__ umask,
                          const float* __restrict__ sent300,
                          float* __restrict__ utt300,
                          _Float16* __restrict__ nodeF)
{
    int bu = blockIdx.x;                   // 0..B*U-1
    int b = bu / U_, u = bu % U_;
    int c = threadIdx.x;
    const int*   ui = utts + (size_t)bu * US_;
    const float* mk = umask + (size_t)bu * US_;
    float cnt = 0.0f;
    for (int k = 0; k < US_; ++k) cnt += mk[k];
    float inv = 1.0f / ((cnt == 0.0f) ? 1.0f : cnt);
    float v = 0.0f;
    if (c < 300) {
        float acc = 0.0f;
        for (int k = 0; k < US_; ++k)
            acc += mk[k] * sent300[((size_t)b * S_ + ui[k]) * 300 + c];
        v = acc * inv;
        utt300[(size_t)bu * 300 + c] = v;
    }
    nodeF[(size_t)(b * PER_ + W_ + I_ + S_ + u) * K300P + c] = (_Float16)v;
}

__global__ void build_sess(const float* __restrict__ utt300,
                           _Float16* __restrict__ nodeF)
{
    int b = blockIdx.x;
    int c = threadIdx.x;
    float v = 0.0f;
    if (c < 300) {
        float acc = 0.0f;
        for (int u = 0; u < U_; ++u)
            acc += utt300[((size_t)b * U_ + u) * 300 + c];
        v = acc * (1.0f / (float)U_);
    }
    nodeF[(size_t)(b * PER_ + PER_ - 1) * K300P + c] = (_Float16)v;
}

__global__ void zero_image_rows(_Float16* __restrict__ nodeF)
{
    int idx = blockIdx.x * blockDim.x + threadIdx.x;
    const int total = B_ * I_ * K300P;
    if (idx >= total) return;
    int row = idx / K300P, c = idx % K300P;
    int b = row / I_, i = row % I_;
    nodeF[(size_t)(b * PER_ + W_ + i) * K300P + c] = (_Float16)0.0f;
}

__global__ void fill_img_rowmap(int* __restrict__ rm)
{
    int i = blockIdx.x * blockDim.x + threadIdx.x;
    if (i < B_ * I_) rm[i] = (i / I_) * PER_ + W_ + (i % I_);
}

// ===================================================================
// CSR-by-dst construction (count -> scan -> scatter)
// ===================================================================
__global__ void zero_int(int* __restrict__ p, int n)
{
    int i = blockIdx.x * blockDim.x + threadIdx.x;
    if (i < n) p[i] = 0;
}

__global__ void count_edges(const int* __restrict__ dst, int* __restrict__ counts, int E)
{
    int e = blockIdx.x * blockDim.x + threadIdx.x;
    if (e < E) atomicAdd(&counts[dst[e]], 1);
}

__global__ void __launch_bounds__(1024)
scan_counts(const int* __restrict__ counts, int* __restrict__ offsets, int n)
{
    __shared__ int part[1024];
    int tid = threadIdx.x;
    int base = tid * 9;
    int loc[9];
    int s = 0;
#pragma unroll
    for (int j = 0; j < 9; ++j) {
        int v = (base + j < n) ? counts[base + j] : 0;
        loc[j] = s;
        s += v;
    }
    part[tid] = s;
    __syncthreads();
    for (int off = 1; off < 1024; off <<= 1) {
        int v = 0;
        if (tid >= off) v = part[tid - off];
        __syncthreads();
        if (tid >= off) part[tid] += v;
        __syncthreads();
    }
    int pre = (tid == 0) ? 0 : part[tid - 1];
#pragma unroll
    for (int j = 0; j < 9; ++j) {
        int idx = base + j;
        if (idx <= n) offsets[idx] = pre + loc[j];
    }
}

__global__ void copy_int(const int* __restrict__ a, int* __restrict__ b, int n)
{
    int i = blockIdx.x * blockDim.x + threadIdx.x;
    if (i < n) b[i] = a[i];
}

__global__ void scatter_edges(const int* __restrict__ src, const int* __restrict__ dst,
                              int* __restrict__ cursor, int* __restrict__ esrc, int E)
{
    int e = blockIdx.x * blockDim.x + threadIdx.x;
    if (e < E) {
        int d = dst[e];
        int pos = atomicAdd(&cursor[d], 1);
        esrc[pos] = src[e];
    }
}

// ===================================================================
// GAT attention logits (f16 features): one wave per (node, head).
// ===================================================================
__global__ void __launch_bounds__(256)
gat_el_er(const _Float16* __restrict__ f, const float* __restrict__ al,
          const float* __restrict__ ar, float* __restrict__ el,
          float* __restrict__ er, int Nnodes, int H, int D)
{
    int wave = threadIdx.x >> 5;
    int lane = threadIdx.x & 31;
    int p = blockIdx.x * 8 + wave;
    if (p >= Nnodes * H) return;
    int node = p / H, h = p % H;
    const _Float16* fr = f + (size_t)node * H * D + (size_t)h * D;
    const float* alr = al + (size_t)h * D;
    const float* arr = ar + (size_t)h * D;
    float sl = 0.0f, sr = 0.0f;
    for (int d = lane; d < D; d += 32) {
        float v = (float)fr[d];
        sl += v * alr[d];
        sr += v * arr[d];
    }
#pragma unroll
    for (int m = 16; m > 0; m >>= 1) {
        sl += __shfl_xor(sl, m, 32);
        sr += __shfl_xor(sr, m, 32);
    }
    if (lane == 0) { el[p] = sl; er[p] = sr; }
}

// ===================================================================
// Edge-softmax + aggregation (f16 in, f16 out), one block per dst node.
// ===================================================================
template <int H, int D, int VD>
__global__ void __launch_bounds__(256)
gat_aggregate(const _Float16* __restrict__ f,    // [N, H*D]
              const int* __restrict__ offsets,   // [N+1]
              const int* __restrict__ esrc,      // [E] src per CSR slot
              const float* __restrict__ el,      // [N,H]
              const float* __restrict__ er,      // [N,H]
              const float* __restrict__ bias,    // [H*D]
              _Float16* __restrict__ out,        // [N, H*D]
              int act)
{
    constexpr int HD = H * D;
    static_assert(HD == 256 * VD, "HD must be 256*VD");
    __shared__ float red[256 * H];
    __shared__ float mx[H], zz[H], ern[H];
    __shared__ int   srcC[64];
    __shared__ float alC[64 * H];

    const int node  = blockIdx.x;
    const int tid   = threadIdx.x;
    const int start = offsets[node];
    const int end   = offsets[node + 1];

    if (tid < H) ern[tid] = er[(size_t)node * H + tid];
    __syncthreads();

    // ---- pass 1: per-head max of leaky_relu(el[src]+er[node]) ----
    float lm[H];
#pragma unroll
    for (int h = 0; h < H; ++h) lm[h] = -1e30f;
    for (int e = start + tid; e < end; e += 256) {
        int s = esrc[e];
#pragma unroll
        for (int h = 0; h < H; ++h) {
            float x = el[(size_t)s * H + h] + ern[h];
            x = (x > 0.0f) ? x : 0.2f * x;
            lm[h] = fmaxf(lm[h], x);
        }
    }
#pragma unroll
    for (int h = 0; h < H; ++h) red[tid * H + h] = lm[h];
    __syncthreads();
    for (int st = 128; st > 0; st >>= 1) {
        if (tid < st) {
#pragma unroll
            for (int h = 0; h < H; ++h)
                red[tid * H + h] = fmaxf(red[tid * H + h], red[(tid + st) * H + h]);
        }
        __syncthreads();
    }
    if (tid < H) mx[tid] = red[tid];
    __syncthreads();

    // ---- pass 2: per-head sum of exp(x - max) ----
    float ls[H];
#pragma unroll
    for (int h = 0; h < H; ++h) ls[h] = 0.0f;
    for (int e = start + tid; e < end; e += 256) {
        int s = esrc[e];
#pragma unroll
        for (int h = 0; h < H; ++h) {
            float x = el[(size_t)s * H + h] + ern[h];
            x = (x > 0.0f) ? x : 0.2f * x;
            ls[h] += __expf(x - mx[h]);
        }
    }
#pragma unroll
    for (int h = 0; h < H; ++h) red[tid * H + h] = ls[h];
    __syncthreads();
    for (int st = 128; st > 0; st >>= 1) {
        if (tid < st) {
#pragma unroll
            for (int h = 0; h < H; ++h)
                red[tid * H + h] += red[(tid + st) * H + h];
        }
        __syncthreads();
    }
    if (tid < H) zz[tid] = red[tid];
    __syncthreads();

    // ---- pass 3: aggregate in chunks of 64 edges ----
    float acc[VD];
#pragma unroll
    for (int v = 0; v < VD; ++v) acc[v] = 0.0f;

    for (int cb = start; cb < end; cb += 64) {
        if (tid < 64) srcC[tid] = (cb + tid < end) ? esrc[cb + tid] : -1;
        __syncthreads();
        if (H == 4) {
            int eL = tid >> 2, h = tid & 3;
            int s = srcC[eL];
            float a = 0.0f;
            if (s >= 0) {
                float x = el[(size_t)s * H + h] + ern[h];
                x = (x > 0.0f) ? x : 0.2f * x;
                a = __expf(x - mx[h]);
            }
            alC[eL * H + h] = a;
        } else {
            if (tid < 64) {
                int s = srcC[tid];
                float a = 0.0f;
                if (s >= 0) {
                    float x = el[s] + ern[0];
                    x = (x > 0.0f) ? x : 0.2f * x;
                    a = __expf(x - mx[0]);
                }
                alC[tid] = a;
            }
        }
        __syncthreads();
        int nInC = end - cb; if (nInC > 64) nInC = 64;
        for (int eL = 0; eL < nInC; ++eL) {
            int s = srcC[eL];
            const _Float16* fr = f + (size_t)s * HD + tid * VD;
#pragma unroll
            for (int v = 0; v < VD; ++v) {
                int d = tid * VD + v;
                int h = d / D;
                acc[v] += alC[eL * H + h] * (float)fr[v];
            }
        }
        __syncthreads();
    }

    // ---- normalize, bias, activation, store ----
#pragma unroll
    for (int v = 0; v < VD; ++v) {
        int d = tid * VD + v;
        int h = d / D;
        float o = acc[v] / zz[h] + bias[d];
        if (act) o = (o > 0.0f) ? o : (__expf(o) - 1.0f);
        out[(size_t)node * HD + d] = (_Float16)o;
    }
}

// ===================================================================
// Final session gather (f16 -> fp32 output)
// ===================================================================
__global__ void gather_sessions(const _Float16* __restrict__ h,
                                const int* __restrict__ sids,
                                float* __restrict__ out)
{
    int idx = blockIdx.x * blockDim.x + threadIdx.x;
    if (idx >= B_ * OUT_) return;
    int b = idx / OUT_, j = idx % OUT_;
    out[idx] = (float)h[(size_t)sids[b] * OUT_ + j];
}

// ===================================================================
// Host-side launch
// ===================================================================
extern "C" void kernel_launch(void* const* d_in, const int* in_sizes, int n_in,
                              void* d_out, int out_size, void* d_ws, size_t ws_size,
                              hipStream_t stream)
{
    const int*   all_words      = (const int*)  d_in[0];
    const float* image_feats    = (const float*)d_in[1];
    const int*   sentences      = (const int*)  d_in[2];
    const float* sentence_mask  = (const float*)d_in[3];
    const int*   utterances     = (const int*)  d_in[4];
    const float* utterance_mask = (const float*)d_in[5];
    const int*   session_ids    = (const int*)  d_in[6];
    const int*   edge_src       = (const int*)  d_in[7];
    const int*   edge_dst       = (const int*)  d_in[8];
    const float* word_embed     = (const float*)d_in[9];
    const float* text_fc_w      = (const float*)d_in[10];
    const float* text_fc_b      = (const float*)d_in[11];
    const float* image_fc_w     = (const float*)d_in[12];
    const float* image_fc_b     = (const float*)d_in[13];
    const float* gat0_fc = (const float*)d_in[14];
    const float* gat0_al = (const float*)d_in[15];
    const float* gat0_ar = (const float*)d_in[16];
    const float* gat0_b  = (const float*)d_in[17];
    const float* gat1_fc = (const float*)d_in[18];
    const float* gat1_al = (const float*)d_in[19];
    const float* gat1_ar = (const float*)d_in[20];
    const float* gat1_b  = (const float*)d_in[21];
    const float* gat2_fc = (const float*)d_in[22];
    const float* gat2_al = (const float*)d_in[23];
    const float* gat2_ar = (const float*)d_in[24];
    const float* gat2_b  = (const float*)d_in[25];
    float* out = (float*)d_out;

    const int E = in_sizes[7];   // 146948

    // ---- workspace carve-out ----
    char* ws = (char*)d_ws;
    size_t off = 0;
    auto carve = [&](size_t bytes) -> char* {
        char* p = ws + off;
        off += (bytes + 255) & ~(size_t)255;
        return p;
    };
    _Float16* nodeF  = (_Float16*)carve((size_t)NNPAD * K300P * 2);
    _Float16* imgF   = (_Float16*)carve((size_t)B_ * I_ * 2048 * 2);
    _Float16* h16A   = (_Float16*)carve((size_t)NNPAD * 1024 * 2);
    _Float16* h16B   = (_Float16*)carve((size_t)NNPAD * 1024 * 2);
    _Float16* f16buf = (_Float16*)carve((size_t)NN_ * 1024 * 2);
    _Float16* wtText = (_Float16*)carve((size_t)1024 * K300P * 2);
    _Float16* wtImg  = (_Float16*)carve((size_t)1024 * 2048 * 2);
    _Float16* wt0    = (_Float16*)carve((size_t)1024 * 1024 * 2);
    _Float16* wt1    = (_Float16*)carve((size_t)1024 * 1024 * 2);
    _Float16* wt2    = (_Float16*)carve((size_t)512 * 1024 * 2);
    float* sent300 = (float*)carve((size_t)B_ * S_ * 300 * 4);
    float* utt300  = (float*)carve((size_t)B_ * U_ * 300 * 4);
    float* elb     = (float*)carve((size_t)NN_ * H_ * 4);
    float* erb     = (float*)carve((size_t)NN_ * H_ * 4);
    int*   counts  = (int*)  carve((size_t)(NN_ + 1) * 4);
    int*   offs    = (int*)  carve((size_t)(NN_ + 1) * 4);
    int*   cursor  = (int*)  carve((size_t)(NN_ + 1) * 4);
    int*   esrc    = (int*)  carve((size_t)E * 4);
    int*   rowmap  = (int*)  carve((size_t)B_ * I_ * 4);
    (void)ws_size;

    // ---- 1) weights: fp32 -> f16 transposed [N][Kpad]; image feats -> f16 ----
    conv_weights<<<(1024 * K300P + 255) / 256, 256, 0, stream>>>(text_fc_w, wtText, 300, K300P, 1024);
    conv_weights<<<(1024 * 2048 + 255) / 256, 256, 0, stream>>>(image_fc_w, wtImg, 2048, 2048, 1024);
    conv_weights<<<(1024 * 1024 + 255) / 256, 256, 0, stream>>>(gat0_fc, wt0, 1024, 1024, 1024);
    conv_weights<<<(1024 * 1024 + 255) / 256, 256, 0, stream>>>(gat1_fc, wt1, 1024, 1024, 1024);
    conv_weights<<<(512 * 1024 + 255) / 256, 256, 0, stream>>>(gat2_fc, wt2, 1024, 1024, 512);
    conv_f32_to_f16<<<(B_ * I_ * 2048 + 255) / 256, 256, 0, stream>>>(image_feats, imgF, B_ * I_ * 2048);

    // ---- 2) feature construction (f16 node matrix) ----
    {
        int tot = B_ * W_ * K300P;
        build_words<<<(tot + 255) / 256, 256, 0, stream>>>(all_words, word_embed, nodeF);
    }
    build_sent<<<B_ * S_, K300P, 0, stream>>>(sentences, sentence_mask, word_embed, sent300, nodeF);
    build_utt<<<B_ * U_, K300P, 0, stream>>>(utterances, utterance_mask, sent300, utt300, nodeF);
    build_sess<<<B_, K300P, 0, stream>>>(utt300, nodeF);
    {
        int tot = B_ * I_ * K300P;
        zero_image_rows<<<(tot + 255) / 256, 256, 0, stream>>>(nodeF);
    }
    fill_img_rowmap<<<1, 256, 0, stream>>>(rowmap);

    // ---- 3) CSR build (graph is fixed across layers) ----
    zero_int<<<(NN_ + 255) / 256, 256, 0, stream>>>(counts, NN_);
    count_edges<<<(E + 255) / 256, 256, 0, stream>>>(edge_dst, counts, E);
    scan_counts<<<1, 1024, 0, stream>>>(counts, offs, NN_);
    copy_int<<<(NN_ + 255) / 256, 256, 0, stream>>>(offs, cursor, NN_);
    scatter_edges<<<(E + 255) / 256, 256, 0, stream>>>(edge_src, edge_dst, cursor, esrc, E);

    const int MB = NNPAD / 128;   // 68 M-blocks (exact)

    // ---- 4) input projections -> h16A [N,1024] ----
    gemm_f16_wmma<<<dim3(MB, 1024 / 64), 256, 0, stream>>>(
        nodeF, K300P, wtText, text_fc_b, h16A, 1024, nullptr, NN_, 1024, K300P);
    gemm_f16_wmma<<<dim3((B_ * I_) / 128, 1024 / 64), 256, 0, stream>>>(
        imgF, 2048, wtImg, image_fc_b, h16A, 1024, rowmap, B_ * I_, 1024, 2048);

    // ---- 5) GAT layer 0: h16A -> h16B ----
    gemm_f16_wmma<<<dim3(MB, 1024 / 64), 256, 0, stream>>>(
        h16A, 1024, wt0, nullptr, f16buf, 1024, nullptr, NN_, 1024, 1024);
    gat_el_er<<<(NN_ * H_ + 7) / 8, 256, 0, stream>>>(f16buf, gat0_al, gat0_ar,
                                                      elb, erb, NN_, H_, D_);
    gat_aggregate<H_, D_, 4><<<NN_, 256, 0, stream>>>(f16buf, offs, esrc, elb, erb,
                                                      gat0_b, h16B, /*act=*/1);

    // ---- 6) GAT layer 1: h16B -> h16A ----
    gemm_f16_wmma<<<dim3(MB, 1024 / 64), 256, 0, stream>>>(
        h16B, 1024, wt1, nullptr, f16buf, 1024, nullptr, NN_, 1024, 1024);
    gat_el_er<<<(NN_ * H_ + 7) / 8, 256, 0, stream>>>(f16buf, gat1_al, gat1_ar,
                                                      elb, erb, NN_, H_, D_);
    gat_aggregate<H_, D_, 4><<<NN_, 256, 0, stream>>>(f16buf, offs, esrc, elb, erb,
                                                      gat1_b, h16A, /*act=*/1);

    // ---- 7) GAT layer 2 (H=1, D=512): h16A -> h16B as dense [N,512] ----
    gemm_f16_wmma<<<dim3(MB, OUT_ / 64), 256, 0, stream>>>(
        h16A, 1024, wt2, nullptr, f16buf, OUT_, nullptr, NN_, OUT_, 1024);
    gat_el_er<<<(NN_ + 7) / 8, 256, 0, stream>>>(f16buf, gat2_al, gat2_ar,
                                                 elb, erb, NN_, 1, OUT_);
    gat_aggregate<1, OUT_, 2><<<NN_, 256, 0, stream>>>(f16buf, offs, esrc, elb, erb,
                                                       gat2_b, h16B, /*act=*/0);

    // ---- 8) gather session rows -> out [B, 512] ----
    gather_sessions<<<(B_ * OUT_ + 255) / 256, 256, 0, stream>>>(h16B, session_ids, out);
}
